// StructPredictionNet_34617436405890
// MI455X (gfx1250) — matrix operand order repaired and checked by reading, not compiled, not run on hardware
//
#include <hip/hip_runtime.h>
#include <hip/hip_bf16.h>

// ---------------------------------------------------------------------------
// StructPredictionNet (3-level 3D U-Net, fp32 reference) for MI455X / gfx1250.
//
// Roofline: ~6.3e10 conv MACs, ~350MB inter-stage traffic (~15us @ 23.3TB/s;
// each stage fits in the 192MB L2) -> compute-bound on the matrix path.
// Convs are implicit GEMM on v_wmma_f32_16x16x32_bf16 (bf16 in / f32 acc).
//
// Steady state per K chunk (verified in round-4 asm): 4x ds_load_b128 (A,
// double-buffered LDS, one wave barrier), s_clause of 4x global_load_b128
// (B, from bf16-prepped weights), 4x v_wmma back-to-back.
//
// Round-5 refinement: all tensor offsets that depend on runtime S3/Cout are
// 32-bit shift/add (they fit in 31 bits; S is a power of two) -> no more
// v_mul_u64 chains in the staging paths. B pointer advances incrementally.
// ---------------------------------------------------------------------------

typedef __attribute__((ext_vector_type(16))) __bf16 v16bf;
typedef __attribute__((ext_vector_type(8)))  __bf16 v8bf;
typedef __attribute__((ext_vector_type(8)))  float  v8f;

#define ROWP 40   // LDS row pitch in bf16 elements (80B; 16B-aligned runs)
#define TILE (16 * ROWP)

// ---------------------------------------------------------------------------
// GroupNorm statistics -> per-channel affine.
// ---------------------------------------------------------------------------
__global__ __launch_bounds__(256)
void gn_stats_kernel(const float* __restrict__ x,
                     const float* __restrict__ gamma,
                     const float* __restrict__ beta,
                     float* __restrict__ ch_scale,
                     float* __restrict__ ch_bias,
                     int C, int S3)
{
    const int g      = blockIdx.x;
    const int groups = gridDim.x;
    const int cpg    = C / groups;
    const size_t n   = (size_t)cpg * (size_t)S3;
    const float* xg  = x + (size_t)g * n;

    float sum = 0.f, ssq = 0.f;
    for (size_t i = threadIdx.x; i < n; i += 256) {
        float v = xg[i];
        sum += v;
        ssq += v * v;
    }
    __shared__ float s1[256];
    __shared__ float s2[256];
    s1[threadIdx.x] = sum;
    s2[threadIdx.x] = ssq;
    __syncthreads();
    for (int off = 128; off > 0; off >>= 1) {
        if ((int)threadIdx.x < off) {
            s1[threadIdx.x] += s1[threadIdx.x + off];
            s2[threadIdx.x] += s2[threadIdx.x + off];
        }
        __syncthreads();
    }
    __shared__ float mv[2];
    if (threadIdx.x == 0) {
        float inv_n = 1.f / (float)n;
        float m   = s1[0] * inv_n;
        float var = s2[0] * inv_n - m * m;
        mv[0] = m;
        mv[1] = rsqrtf(var + 1e-5f);
    }
    __syncthreads();
    if ((int)threadIdx.x < cpg) {
        int c = g * cpg + threadIdx.x;
        float sc = gamma[c] * mv[1];
        ch_scale[c] = sc;
        ch_bias[c]  = beta[c] - mv[0] * sc;
    }
}

// Identity affine for the pre-conv.
__global__ __launch_bounds__(256)
void affine_identity_kernel(float* __restrict__ scl, float* __restrict__ sb)
{
    scl[threadIdx.x] = 1.f;
    sb[threadIdx.x]  = 0.f;
}

// ---------------------------------------------------------------------------
// Weight prep: w[Cout][Cin][27] f32 -> wt2[t][ch][oc][kk] bf16, kk zero-padded
// to 32. Cold kernel (<=332K elements); divisions are fine here.
// ---------------------------------------------------------------------------
__global__ __launch_bounds__(256)
void wprep_kernel(const float* __restrict__ w, __bf16* __restrict__ wt2,
                  int Cin, int Cout, int chunks)
{
    size_t total = (size_t)27 * chunks * Cout * 32;
    size_t i = (size_t)blockIdx.x * 256 + threadIdx.x;
    if (i >= total) return;
    int kk = (int)(i & 31);
    size_t r  = i >> 5;
    int oc = (int)(r % Cout);
    size_t r2 = r / Cout;
    int ch = (int)(r2 % chunks);
    int t  = (int)(r2 / chunks);
    int c  = ch * 32 + kk;
    float v = (c < Cin) ? w[((size_t)oc * Cin + c) * 27 + t] : 0.f;
    wt2[i] = (__bf16)v;
}

// ---------------------------------------------------------------------------
// Branchless im2col sample: clamped (safe) address, always load, select 0 for
// OOB, apply per-channel affine. All offsets are 32-bit shift/add (c*S3 and
// friends fit in 31 bits for every layer of this net).
// ---------------------------------------------------------------------------
__device__ __forceinline__
float im2col_sample(const float* __restrict__ in,
                    const float* __restrict__ sc, const float* __restrict__ sb,
                    int c, int vox, int kd, int kh, int kw, int ls, int S)
{
    int z = vox >> (2 * ls);
    int y = (vox >> ls) & (S - 1);
    int x = vox & (S - 1);
    int zz = z + kd - 1, yy = y + kh - 1, xx = x + kw - 1;
    bool inb = ((unsigned)zz < (unsigned)S) & ((unsigned)yy < (unsigned)S) &
               ((unsigned)xx < (unsigned)S);
    int zc = zz < 0 ? 0 : (zz >= S ? S - 1 : zz);
    int yc = yy < 0 ? 0 : (yy >= S ? S - 1 : yy);
    int xc = xx < 0 ? 0 : (xx >= S ? S - 1 : xx);
    int off = (c << (3 * ls)) + (zc << (2 * ls)) + (yc << ls) + xc;  // 32-bit
    float v = in[off];
    v = fmaf(v, sc[c], sb[c]);
    return inb ? v : 0.f;
}

// ---------------------------------------------------------------------------
// Implicit-GEMM 3x3x3 SAME conv via WMMA bf16, 2x2 register tiling.
//   Block: (32, NW). Per block: 32 output voxels (2 M tiles via LDS).
//   Per wave: 32 output channels (2 N tiles direct from wt2 in global).
//   4 accumulators -> 4 wmma per K chunk; one barrier per chunk (A double-
//   buffered). K order: tap-major; each 32-chunk = one tap x <=32 channels.
// Requires: S3 % 32 == 0, Cout % 32 == 0.
// ---------------------------------------------------------------------------
template <int NW>
__global__ __launch_bounds__(32 * NW)
void conv3_wmma_kernel(const float* __restrict__ in,
                       const __bf16* __restrict__ wt2,     // [27][ch][Cout][32]
                       const float* __restrict__ in_scale, // per-channel affine
                       const float* __restrict__ in_bias,
                       const float* __restrict__ out_bias, // nullable
                       float* __restrict__ out,
                       int Cin, int Cout, int chunks, int ls, int relu)
{
    const int S  = 1 << ls;
    const int lane = threadIdx.x;
    const int wv   = threadIdx.y;
    const int tid  = wv * 32 + lane;

    const int mbase  = blockIdx.x * 32;                    // 32 voxels / block
    const int ntile0 = (blockIdx.y * NW + wv) * 2;         // 2 N tiles / wave

    __shared__ __align__(16) __bf16 Asm[2 * 2 * TILE];     // double-buffered

    const int half = lane >> 4;
    const int nn   = lane & 15;

    // Per-lane B fragment pointer; advances by Cout*32 bf16 per K chunk.
    const __bf16* bp = wt2 + ((ntile0 * 16 + nn) << 5) + half * 8;
    const int bstep = Cout << 5;

    v8f acc00 = {}, acc01 = {}, acc10 = {}, acc11 = {};
    int pq = 0;

    #pragma unroll 1
    for (int t = 0; t < 27; ++t) {
        const int kd = t / 9;                  // uniform (SALU), per tap
        const int kh = (t - kd * 9) / 3;
        const int kw = t - kd * 9 - kh * 3;

        #pragma unroll 1
        for (int ch = 0; ch < chunks; ++ch, bp += bstep) {
            const int c0 = ch * 32;
            const int cW = (Cin - c0 < 32) ? (Cin - c0) : 32;

            __bf16* Ab = &Asm[pq ? 2 * TILE : 0];
            pq ^= 1;

            // ---- stage A: 1024 elements, voxel-fastest (coalesced) ----
            if (cW == 32) {                    // predicate-free common path
                #pragma unroll
                for (int it = 0; it < 1024 / (32 * NW); ++it) {
                    int idx = it * (32 * NW) + tid;
                    int m   = idx & 31;
                    int kk  = idx >> 5;
                    float val = im2col_sample(in, in_scale, in_bias,
                                              c0 + kk, mbase + m, kd, kh, kw,
                                              ls, S);
                    Ab[(m >> 4) * TILE + (m & 15) * ROWP + kk] = (__bf16)val;
                }
            } else {                           // pre-conv only (Cin=3)
                #pragma unroll
                for (int it = 0; it < 1024 / (32 * NW); ++it) {
                    int idx = it * (32 * NW) + tid;
                    int m   = idx & 31;
                    int kk  = idx >> 5;
                    int ckk = kk < cW ? kk : 0;        // safe channel
                    float val = im2col_sample(in, in_scale, in_bias,
                                              c0 + ckk, mbase + m, kd, kh, kw,
                                              ls, S);
                    Ab[(m >> 4) * TILE + (m & 15) * ROWP + kk] =
                        (__bf16)(kk < cW ? val : 0.f);
                }
            }
            __syncthreads();

            // ---- A fragments: 2x ds_load_b128 each (16B-aligned runs) ----
            const __bf16* ar0 = &Ab[nn * ROWP + half * 8];
            const __bf16* ar1 = ar0 + TILE;
            v8bf a0lo = *(const v8bf*)ar0;
            v8bf a0hi = *(const v8bf*)(ar0 + 16);
            v8bf a1lo = *(const v8bf*)ar1;
            v8bf a1hi = *(const v8bf*)(ar1 + 16);
            v16bf a0 = __builtin_shufflevector(a0lo, a0hi,
                        0,1,2,3,4,5,6,7,8,9,10,11,12,13,14,15);
            v16bf a1 = __builtin_shufflevector(a1lo, a1hi,
                        0,1,2,3,4,5,6,7,8,9,10,11,12,13,14,15);

            // ---- B fragments: 2x global_load_b128 each, from wt2 ----
            v8bf b0lo = *(const v8bf*)bp;
            v8bf b0hi = *(const v8bf*)(bp + 16);
            v8bf b1lo = *(const v8bf*)(bp + 16 * 32);   // oc + 16 rows
            v8bf b1hi = *(const v8bf*)(bp + 16 * 32 + 16);
            v16bf b0 = __builtin_shufflevector(b0lo, b0hi,
                        0,1,2,3,4,5,6,7,8,9,10,11,12,13,14,15);
            v16bf b1 = __builtin_shufflevector(b1lo, b1hi,
                        0,1,2,3,4,5,6,7,8,9,10,11,12,13,14,15);

            acc00 = __builtin_amdgcn_wmma_f32_16x16x32_bf16(false, a0, false, b0,
                                                            (short)0, acc00, false, false);
            acc01 = __builtin_amdgcn_wmma_f32_16x16x32_bf16(false, a0, false, b1,
                                                            (short)0, acc01, false, false);
            acc10 = __builtin_amdgcn_wmma_f32_16x16x32_bf16(false, a1, false, b0,
                                                            (short)0, acc10, false, false);
            acc11 = __builtin_amdgcn_wmma_f32_16x16x32_bf16(false, a1, false, b1,
                                                            (short)0, acc11, false, false);
        }
    }

    // ---- write D tiles: lane n = lane%16, row m = r + (lane/16)*8 ----
    const int oc0   = ntile0 * 16 + nn;
    const int oc1   = oc0 + 16;
    const int obase0 = oc0 << (3 * ls);       // 32-bit channel offsets
    const int obase1 = oc1 << (3 * ls);
    const int half8 = half * 8;
    const float ob0 = out_bias ? out_bias[oc0] : 0.f;
    const float ob1 = out_bias ? out_bias[oc1] : 0.f;
    #pragma unroll
    for (int r = 0; r < 8; ++r) {
        int vox0 = mbase + r + half8;
        int vox1 = vox0 + 16;
        float v;
        v = acc00[r] + ob0; if (relu) v = fmaxf(v, 0.f);
        out[obase0 + vox0] = v;
        v = acc01[r] + ob1; if (relu) v = fmaxf(v, 0.f);
        out[obase1 + vox0] = v;
        v = acc10[r] + ob0; if (relu) v = fmaxf(v, 0.f);
        out[obase0 + vox1] = v;
        v = acc11[r] + ob1; if (relu) v = fmaxf(v, 0.f);
        out[obase1 + vox1] = v;
    }
}

// ---------------------------------------------------------------------------
// Direct conv for the 2-channel struct heads (N too narrow for a WMMA tile).
// ---------------------------------------------------------------------------
__global__ __launch_bounds__(256)
void conv3_small_kernel(const float* __restrict__ in,
                        const float* __restrict__ w,   // [2, Cin, 27]
                        const float* __restrict__ b,   // [2]
                        float* __restrict__ out,       // [2, S^3]
                        int Cin, int ls)
{
    const int S  = 1 << ls;
    const int S3 = 1 << (3 * ls);
    int vox = blockIdx.x * blockDim.x + threadIdx.x;
    if (vox >= S3) return;
    int z = vox >> (2 * ls);
    int y = (vox >> ls) & (S - 1);
    int x = vox & (S - 1);

    float acc0 = b[0], acc1 = b[1];
    for (int c = 0; c < Cin; ++c) {
        const float* ip  = in + (c << (3 * ls));
        const float* wp0 = w + c * 27;
        const float* wp1 = w + (Cin + c) * 27;
        int t = 0;
        #pragma unroll
        for (int kd = -1; kd <= 1; ++kd) {
            int zz = z + kd;
            #pragma unroll
            for (int kh = -1; kh <= 1; ++kh) {
                int yy = y + kh;
                #pragma unroll
                for (int kw = -1; kw <= 1; ++kw, ++t) {
                    int xx = x + kw;
                    if ((unsigned)zz < (unsigned)S && (unsigned)yy < (unsigned)S &&
                        (unsigned)xx < (unsigned)S) {
                        float v = ip[(zz << (2 * ls)) + (yy << ls) + xx];
                        acc0 = fmaf(v, wp0[t], acc0);
                        acc1 = fmaf(v, wp1[t], acc1);
                    }
                }
            }
        }
    }
    out[vox]      = acc0;
    out[S3 + vox] = acc1;
}

// ---------------------------------------------------------------------------
// 2x2x2 max pool, stride 2 (lso = log2 of output size).
// ---------------------------------------------------------------------------
__global__ __launch_bounds__(256)
void maxpool2_kernel(const float* __restrict__ in, float* __restrict__ out,
                     int C, int lso)
{
    const int So  = 1 << lso;
    const int So3 = 1 << (3 * lso);
    const int lsi = lso + 1;
    int i = blockIdx.x * blockDim.x + threadIdx.x;
    if (i >= (C << (3 * lso))) return;
    int c = i >> (3 * lso);
    int r = i & (So3 - 1);
    int z = r >> (2 * lso);
    int y = (r >> lso) & (So - 1);
    int x = r & (So - 1);
    const float* ip = in + (c << (3 * lsi));
    float m = -__builtin_inff();
    #pragma unroll
    for (int dz = 0; dz < 2; ++dz)
        #pragma unroll
        for (int dy = 0; dy < 2; ++dy)
            #pragma unroll
            for (int dx = 0; dx < 2; ++dx) {
                float v = ip[((2 * z + dz) << (2 * lsi)) + ((2 * y + dy) << lsi) + (2 * x + dx)];
                m = fmaxf(m, v);
            }
    out[i] = m;
}

// ---------------------------------------------------------------------------
// Mask-gated x2 nearest upsample into low channels of a concat buffer.
// ---------------------------------------------------------------------------
__global__ __launch_bounds__(256)
void up2_gate_kernel(const float* __restrict__ in, const float* __restrict__ s,
                     float* __restrict__ out, int C, int lsi)
{
    const int lso = lsi + 1;
    const int So  = 1 << lso;
    const int Si3 = 1 << (3 * lsi);
    size_t i = (size_t)blockIdx.x * blockDim.x + threadIdx.x;
    if (i >= ((size_t)C << (3 * lso))) return;
    int c = (int)(i >> (3 * lso));
    int r = (int)(i & ((1u << (3 * lso)) - 1));
    int z = r >> (2 * lso);
    int y = (r >> lso) & (So - 1);
    int x = r & (So - 1);
    int vi = ((z >> 1) << (2 * lsi)) + ((y >> 1) << lsi) + (x >> 1);
    float m = (s[vi] > s[Si3 + vi]) ? 1.f : 0.f;
    out[i] = in[(c << (3 * lsi)) + vi] * m;
}

// ---------------------------------------------------------------------------
// In-place mask gate at same resolution.
// ---------------------------------------------------------------------------
__global__ __launch_bounds__(256)
void gate_inplace_kernel(float* __restrict__ x, const float* __restrict__ s,
                         int C, int ls3)
{
    size_t i = (size_t)blockIdx.x * blockDim.x + threadIdx.x;
    if (i >= ((size_t)C << ls3)) return;
    int vi = (int)(i & ((1u << ls3) - 1));
    float m = (s[vi] > s[(1 << ls3) + vi]) ? 1.f : 0.f;
    x[i] *= m;
}

// ---------------------------------------------------------------------------
// Host-side orchestration.
// ---------------------------------------------------------------------------
static inline void run_gn(const float* x, const float* gamma, const float* beta,
                          float* scl, float* sb, int C, int S3, hipStream_t st)
{
    gn_stats_kernel<<<8, 256, 0, st>>>(x, gamma, beta, scl, sb, C, S3);
}

static inline void run_conv(const float* in, const float* w, __bf16* wt2,
                            const float* scl, const float* sb,
                            const float* outb, float* out,
                            int Cin, int Cout, int ls, int relu, hipStream_t st)
{
    int chunks = (Cin + 31) / 32;
    size_t wtot = (size_t)27 * chunks * Cout * 32;
    wprep_kernel<<<(unsigned)((wtot + 255) / 256), 256, 0, st>>>(w, wt2, Cin, Cout, chunks);

    int pairs = Cout / 32;                // N-tile pairs (2x16 channels each)
    int nw    = pairs < 4 ? pairs : 4;    // waves per block
    int S3    = 1 << (3 * ls);
    dim3 grid((unsigned)(S3 / 32), (unsigned)(pairs / nw));
    dim3 block(32, (unsigned)nw);
    switch (nw) {
    case 1: conv3_wmma_kernel<1><<<grid, block, 0, st>>>(in, wt2, scl, sb, outb, out, Cin, Cout, chunks, ls, relu); break;
    case 2: conv3_wmma_kernel<2><<<grid, block, 0, st>>>(in, wt2, scl, sb, outb, out, Cin, Cout, chunks, ls, relu); break;
    default: conv3_wmma_kernel<4><<<grid, block, 0, st>>>(in, wt2, scl, sb, outb, out, Cin, Cout, chunks, ls, relu); break;
    }
}

extern "C" void kernel_launch(void* const* d_in, const int* in_sizes, int n_in,
                              void* d_out, int out_size, void* d_ws, size_t ws_size,
                              hipStream_t stream)
{
    (void)in_sizes; (void)n_in; (void)out_size; (void)ws_size;

    // Input order: x, then params leaves in _make_params insertion order.
    const float* X = (const float*)d_in[0];
    #define P(i) ((const float*)d_in[(i)])
    // 1 pre_w, 2 pre_b
    // 3..8   d1  (gs,gb,w)x2      9..14  d2     15..20 d3
    // 21..26 dd2                  27..32 dd1
    // 33 s3_w, 34 s3_b, 35 s2_w, 36 s2_b, 37 s1_w, 38 s1_b

    const size_t V64 = 262144, V32 = 32768, V16 = 4096;
    float* ws   = (float*)d_ws;
    float* enc1 = ws;                          // 32*V64   (skip, long-lived)
    float* R1   = enc1 + 32 * V64;             // 96*V64   (bufA/bufB, later concat1)
    float* bufA = R1;
    float* bufB = R1 + 32 * V64;
    float* enc2 = R1 + 96 * V64;               // 64*V32   (skip, later dd2 out)
    float* Preg = enc2 + 64 * V32;             // 64*V32   (P1/P2, later dd2 tmp)
    float* P1   = Preg;
    float* P2   = Preg + 32 * V32;
    float* Q1   = Preg + 64 * V32;             // 64*V16
    float* Q2   = Q1 + 64 * V16;               // 64*V16
    float* Q3   = Q2 + 64 * V16;               // 128*V16
    float* cat2 = Q3 + 128 * V16;              // 192*V32
    float* dd1t = cat2 + 192 * V32;            // 32*V64
    float* scl  = dd1t + 32 * V64;             // 256
    float* sb   = scl + 256;                   // 256
    __bf16* wt2 = (__bf16*)(sb + 256);         // up to 27*6*64*32 bf16 (~648KB)

    float* outX  = (float*)d_out;              // [32, 64^3]
    float* outS0 = outX + 32 * V64;            // [2, 64^3]
    float* outS1 = outS0 + 2 * V64;            // [2, 32^3]
    float* outS2 = outS1 + 2 * V32;            // [2, 16^3]

    // ---- pre conv: 3 -> 32 @64^3, identity affine, +bias, no ReLU ----
    affine_identity_kernel<<<1, 256, 0, stream>>>(scl, sb);
    run_conv(X, P(1), wt2, scl, sb, P(2), bufA, 3, 32, 6, 0, stream);

    // ---- d1: (GN,conv32->32,ReLU) x2 ----
    run_gn(bufA, P(3), P(4), scl, sb, 32, (int)V64, stream);
    run_conv(bufA, P(5), wt2, scl, sb, nullptr, bufB, 32, 32, 6, 1, stream);
    run_gn(bufB, P(6), P(7), scl, sb, 32, (int)V64, stream);
    run_conv(bufB, P(8), wt2, scl, sb, nullptr, enc1, 32, 32, 6, 1, stream);

    // ---- pool -> d2: 32->32, 32->64 @32^3 ----
    maxpool2_kernel<<<(unsigned)((32 * V32 + 255) / 256), 256, 0, stream>>>(enc1, P1, 32, 5);
    run_gn(P1, P(9), P(10), scl, sb, 32, (int)V32, stream);
    run_conv(P1, P(11), wt2, scl, sb, nullptr, P2, 32, 32, 5, 1, stream);
    run_gn(P2, P(12), P(13), scl, sb, 32, (int)V32, stream);
    run_conv(P2, P(14), wt2, scl, sb, nullptr, enc2, 32, 64, 5, 1, stream);

    // ---- pool -> d3: 64->64, 64->128 @16^3 ----
    maxpool2_kernel<<<(unsigned)((64 * V16 + 255) / 256), 256, 0, stream>>>(enc2, Q1, 64, 4);
    run_gn(Q1, P(15), P(16), scl, sb, 64, (int)V16, stream);
    run_conv(Q1, P(17), wt2, scl, sb, nullptr, Q2, 64, 64, 4, 1, stream);
    run_gn(Q2, P(18), P(19), scl, sb, 64, (int)V16, stream);
    run_conv(Q2, P(20), wt2, scl, sb, nullptr, Q3, 64, 128, 4, 1, stream);

    // ---- struct2 head; gated upsample into concat2; append enc2 ----
    conv3_small_kernel<<<(unsigned)((V16 + 255) / 256), 256, 0, stream>>>(Q3, P(33), P(34), outS2, 128, 4);
    up2_gate_kernel<<<(unsigned)((128 * V32 + 255) / 256), 256, 0, stream>>>(Q3, outS2, cat2, 128, 4);
    hipMemcpyAsync(cat2 + 128 * V32, enc2, 64 * V32 * sizeof(float),
                   hipMemcpyDeviceToDevice, stream);

    // ---- dd2: 192->64, 64->64 @32^3 ----
    run_gn(cat2, P(21), P(22), scl, sb, 192, (int)V32, stream);
    run_conv(cat2, P(23), wt2, scl, sb, nullptr, Preg, 192, 64, 5, 1, stream);
    run_gn(Preg, P(24), P(25), scl, sb, 64, (int)V32, stream);
    run_conv(Preg, P(26), wt2, scl, sb, nullptr, enc2, 64, 64, 5, 1, stream);  // enc2 = dd2 out

    // ---- struct1 head; gated upsample into concat1 (R1); append enc1 ----
    conv3_small_kernel<<<(unsigned)((V32 + 255) / 256), 256, 0, stream>>>(enc2, P(35), P(36), outS1, 64, 5);
    up2_gate_kernel<<<(unsigned)((64 * V64 + 255) / 256), 256, 0, stream>>>(enc2, outS1, R1, 64, 5);
    hipMemcpyAsync(R1 + 64 * V64, enc1, 32 * V64 * sizeof(float),
                   hipMemcpyDeviceToDevice, stream);

    // ---- dd1: 96->32, 32->32 @64^3; second conv writes straight to d_out ----
    run_gn(R1, P(27), P(28), scl, sb, 96, (int)V64, stream);
    run_conv(R1, P(29), wt2, scl, sb, nullptr, dd1t, 96, 32, 6, 1, stream);
    run_gn(dd1t, P(30), P(31), scl, sb, 32, (int)V64, stream);
    run_conv(dd1t, P(32), wt2, scl, sb, nullptr, outX, 32, 32, 6, 1, stream);

    // ---- struct0 head; final in-place gate ----
    conv3_small_kernel<<<(unsigned)((V64 + 255) / 256), 256, 0, stream>>>(outX, P(37), P(38), outS0, 32, 6);
    gate_inplace_kernel<<<(unsigned)((32 * V64 + 255) / 256), 256, 0, stream>>>(outX, outS0, 32, 18);

    #undef P
}
